// MoeBlock_48988396978788
// MI455X (gfx1250) — matrix operand order repaired and checked by reading, not compiled, not run on hardware
//
#include <hip/hip_runtime.h>

// ---- problem constants (B=2,S=2048 -> T=4096; H=1024; F=2048; E=8; top-2) ----
#define T_TOK 4096
#define HDIM  1024
#define FDIM  2048
#define NEXP  8

// ---- feature probes (safe: fall back to sync copies if builtins absent) -----
#ifndef __has_builtin
#define __has_builtin(x) 0
#endif
#if defined(__AMDGCN__) && __has_builtin(__builtin_amdgcn_global_load_async_to_lds_b128) && __has_builtin(__builtin_amdgcn_s_wait_asynccnt)
#define USE_ASYNC 1
#else
#define USE_ASYNC 0
#endif
#if defined(__AMDGCN__) && __has_builtin(__builtin_amdgcn_tensor_load_to_lds) && __has_builtin(__builtin_amdgcn_s_wait_tensorcnt)
#define USE_TDM 1
#else
#define USE_TDM 0
#endif

// ---- vector types (own ext_vector types: trivially copyable, bit_cast-safe) ----
typedef __attribute__((ext_vector_type(16))) __bf16       v16bf;
typedef __attribute__((ext_vector_type(8)))  __bf16       v8bf;
typedef __attribute__((ext_vector_type(4)))  __bf16       v4bf;
typedef __attribute__((ext_vector_type(8)))  float        v8f;
typedef __attribute__((ext_vector_type(4)))  float        f32x4;
typedef __attribute__((ext_vector_type(4)))  unsigned int u32x4;
typedef __attribute__((ext_vector_type(8)))  int          i32x8;
typedef __attribute__((ext_vector_type(4)))  int          i32x4;

struct Raw32 { u32x4 lo, hi; };   // 32 bytes == one v16bf fragment

#define LDS_PITCH 40               // bf16/row: 80B pitch, 16B-aligned, bank-conflict-free

// Build A fragment (16x32 bf16, M x K): lane m = lane&15 holds two contiguous
// 8-element K chunks at K = kh and K = 16+kh where kh = (lane>>4)*8.
__device__ __forceinline__ v16bf lds_frag_a(const __bf16* Ls, int row, int lane) {
  const int kh = (lane >> 4) * 8;
  const u32x4* p0 = (const u32x4*)(Ls + row * LDS_PITCH + kh);
  const u32x4* p1 = (const u32x4*)(Ls + row * LDS_PITCH + 16 + kh);
  Raw32 r{*p0, *p1};
  return __builtin_bit_cast(v16bf, r);
}

// Build B fragment (32x16 bf16, K x N): lane n = lane&15 (row n of the [N,K]
// weight tile) holds two contiguous 8-element K chunks at K = kh and K = 16+kh
// where kh = (lane>>4)*8.
__device__ __forceinline__ v16bf lds_frag_b(const __bf16* Ls, int row, int lane) {
  const int kh = (lane >> 4) * 8;
  const u32x4* p0 = (const u32x4*)(Ls + row * LDS_PITCH + kh);
  const u32x4* p1 = (const u32x4*)(Ls + row * LDS_PITCH + 16 + kh);
  Raw32 r{*p0, *p1};
  return __builtin_bit_cast(v16bf, r);
}

__device__ __forceinline__ void cvt16_store(f32x4 f0, f32x4 f1, f32x4 f2, f32x4 f3,
                                            __bf16* dst) {
  v8bf lo, hi;
  lo[0]=(__bf16)f0.x; lo[1]=(__bf16)f0.y; lo[2]=(__bf16)f0.z; lo[3]=(__bf16)f0.w;
  lo[4]=(__bf16)f1.x; lo[5]=(__bf16)f1.y; lo[6]=(__bf16)f1.z; lo[7]=(__bf16)f1.w;
  hi[0]=(__bf16)f2.x; hi[1]=(__bf16)f2.y; hi[2]=(__bf16)f2.z; hi[3]=(__bf16)f2.w;
  hi[4]=(__bf16)f3.x; hi[5]=(__bf16)f3.y; hi[6]=(__bf16)f3.z; hi[7]=(__bf16)f3.w;
  u32x4* d = (u32x4*)dst;
  d[0] = __builtin_bit_cast(u32x4, lo);
  d[1] = __builtin_bit_cast(u32x4, hi);
}

// Per-lane bf16 copy of 32B (global -> LDS), async (ASYNCcnt) if available.
// Builtin signature (from compiler diagnostic): (AS1 v4i32*, AS3 v4i32*, imm, imm)
__device__ __forceinline__ void copy32B_to_lds(const __bf16* g, __bf16* l) {
#if USE_ASYNC
  __builtin_amdgcn_global_load_async_to_lds_b128(
      (__attribute__((address_space(1))) i32x4*)(__attribute__((address_space(1))) void*)(void*)g,
      (__attribute__((address_space(3))) i32x4*)(__attribute__((address_space(3))) void*)(void*)l,
      0, 0);
  __builtin_amdgcn_global_load_async_to_lds_b128(
      (__attribute__((address_space(1))) i32x4*)(__attribute__((address_space(1))) void*)(void*)(g + 8),
      (__attribute__((address_space(3))) i32x4*)(__attribute__((address_space(3))) void*)(void*)(l + 8),
      0, 0);
#else
  const u32x4* s = (const u32x4*)g;
  u32x4 v0 = s[0], v1 = s[1];
  u32x4* d = (u32x4*)l;
  d[0] = v0; d[1] = v1;
#endif
}

#if USE_TDM
// One TDM descriptor moves a whole 128-row x 32-col bf16 tile (row pitch FDIM
// elements in global) into LDS with 16B padding every 64B -> LDS_PITCH layout.
// Issue from a single wave; completion tracked by TENSORcnt.
__device__ __forceinline__ void tdm_load_tile_bf16(const __bf16* gtile, unsigned lds_byte) {
  unsigned long long ga = (unsigned long long)(size_t)gtile;
  u32x4 g0;
  g0[0] = 1u;                                           // count=1, user desc
  g0[1] = lds_byte;                                     // lds_addr
  g0[2] = (unsigned)ga;                                 // global_addr[31:0]
  g0[3] = (unsigned)((ga >> 32) & 0x01FFFFFFu) | (2u << 30); // addr[56:32] | type=2
  i32x8 g1;
  g1[0] = (int)((1u << 16)        // data_size = 2 bytes
              | (1u << 20)        // pad_enable
              | (3u << 22)        // pad_interval: 16 DWORDs (64B of data)
              | (3u << 25));      // pad_amount:   4 DWORDs (16B pad)
  g1[1] = (int)(((unsigned)FDIM & 0xFFFFu) << 16);      // tensor_dim0[15:0]
  g1[2] = (int)((((unsigned)FDIM >> 16) & 0xFFFFu)      // tensor_dim0[31:16]
              | (((unsigned)T_TOK & 0xFFFFu) << 16));   // tensor_dim1[15:0]
  g1[3] = (int)((((unsigned)T_TOK >> 16) & 0xFFFFu)     // tensor_dim1[31:16]
              | (32u << 16));                           // tile_dim0 = 32 elems
  g1[4] = (int)128u;                                    // tile_dim1 = 128 rows
  g1[5] = (int)FDIM;                                    // tensor_dim0_stride[31:0]
  g1[6] = 0;
  g1[7] = 0;
  i32x4 z4; z4[0] = z4[1] = z4[2] = z4[3] = 0;
#if __clang_major__ >= 23
  i32x8 z8; for (int i = 0; i < 8; ++i) z8[i] = 0;
  __builtin_amdgcn_tensor_load_to_lds(g0, g1, z4, z4, z8, 0);
#else
  __builtin_amdgcn_tensor_load_to_lds(g0, g1, z4, z4, 0);
#endif
}
__device__ __forceinline__ unsigned lds_addr_u32(const void* p) {
  return (unsigned)(size_t)(__attribute__((address_space(3))) const void*)p;
}
#endif

// --------------------------- tiny helper kernels -----------------------------
__global__ void zero_f32_kernel(float* __restrict__ p, int n) {
  int i = blockIdx.x * blockDim.x + threadIdx.x;
  if (i < n) p[i] = 0.0f;
}

__global__ void cvt_f32_bf16_kernel(const float* __restrict__ src,
                                    __bf16* __restrict__ dst, int n4) {
  int i = blockIdx.x * blockDim.x + threadIdx.x;
  if (i < n4) {
    f32x4 f = ((const f32x4*)src)[i];
    v4bf b;
    b[0]=(__bf16)f.x; b[1]=(__bf16)f.y; b[2]=(__bf16)f.z; b[3]=(__bf16)f.w;
    ((unsigned long long*)dst)[i] = __builtin_bit_cast(unsigned long long, b);
  }
}

// Router: logits = x @ Wg^T, softmax (max-shifted), top-2, renormalize.
__global__ void router_topk_kernel(const float* __restrict__ x,
                                   const float* __restrict__ wg,
                                   float* __restrict__ wfull) {
  int t = blockIdx.x * blockDim.x + threadIdx.x;
  if (t >= T_TOK) return;
  const float* xr = x + (size_t)t * HDIM;
  float acc[NEXP];
  #pragma unroll
  for (int e = 0; e < NEXP; ++e) acc[e] = 0.0f;
  for (int h = 0; h < HDIM; ++h) {
    float xv = xr[h];
    #pragma unroll
    for (int e = 0; e < NEXP; ++e) acc[e] += xv * wg[e * HDIM + h];
  }
  float m = acc[0];
  #pragma unroll
  for (int e = 1; e < NEXP; ++e) m = fmaxf(m, acc[e]);
  float p[NEXP];
  #pragma unroll
  for (int e = 0; e < NEXP; ++e) p[e] = expf(acc[e] - m);
  int i1 = 0; float b1 = p[0];
  #pragma unroll
  for (int e = 1; e < NEXP; ++e) if (p[e] > b1) { b1 = p[e]; i1 = e; }
  int i2 = (i1 == 0) ? 1 : 0; float b2 = p[i2];
  #pragma unroll
  for (int e = 0; e < NEXP; ++e)
    if (e != i1 && p[e] > b2) { b2 = p[e]; i2 = e; }
  float inv = 1.0f / (b1 + b2);
  #pragma unroll
  for (int e = 0; e < NEXP; ++e) wfull[(size_t)t * NEXP + e] = 0.0f;
  wfull[(size_t)t * NEXP + i1] = b1 * inv;
  wfull[(size_t)t * NEXP + i2] = b2 * inv;
}

// ------------------- GEMM A: h = gelu(x@Wg_e^T) * (x@Wu_e^T) -----------------
// 128 tokens x 64 f-cols per block; 8 waves 2(M)x4(N); each wave owns matched
// gate and up accumulators so the GELU*up fusion stays in-register.
// Double-buffered LDS; A-tile staged via ASYNCcnt path when available.
// All 6 fragments are hoisted ahead of the 8 WMMAs: one ds_load clause, one
// wait, then back-to-back v_wmma (no D->A/B hazard between them).
__global__ __launch_bounds__(256)
void moe_gateup_gelu_kernel(const __bf16* __restrict__ xb,   // [T, H] bf16
                            const float*  __restrict__ wgu,  // [2F, H] f32 (expert e)
                            __bf16* __restrict__ hws)        // [T, F] bf16
{
  __shared__ __bf16 As[2][128 * LDS_PITCH];
  __shared__ __bf16 Bs[2][128 * LDS_PITCH];
  const int fbase = blockIdx.x * 64;
  const int tbase = blockIdx.y * 128;
  const int tid  = threadIdx.x;
  const int lane = tid & 31;
  const int w    = tid >> 5;
  const int wm   = w >> 2;          // 0..1 -> 64-row slab
  const int wn   = w & 3;           // 0..3 -> 16-col slab
  const int r    = tid >> 1;        // staging row 0..127
  const int half = tid & 1;         // staging col half

  v8f accg[4], accu[4];
  #pragma unroll
  for (int i = 0; i < 4; ++i)
    #pragma unroll
    for (int j = 0; j < 8; ++j) { accg[i][j] = 0.0f; accu[i][j] = 0.0f; }

  const int jrow = (r < 64) ? (fbase + r) : (FDIM + fbase + (r - 64));
  const __bf16* agsrc = xb + (size_t)(tbase + r) * HDIM;
  const float*  bgsrc = wgu + (size_t)jrow * HDIM;

  auto stage = [&](int k0, int b) {
    copy32B_to_lds(agsrc + k0 + half * 16, &As[b][r * LDS_PITCH + half * 16]);
    const f32x4* g = (const f32x4*)(bgsrc + k0 + half * 16);
    cvt16_store(g[0], g[1], g[2], g[3], &Bs[b][r * LDS_PITCH + half * 16]);
    if (k0 + 32 < HDIM) __builtin_prefetch(bgsrc + k0 + 32, 0, 1);
  };
  auto pipe_sync = [&]() {
#if USE_ASYNC
    __builtin_amdgcn_s_wait_asynccnt(0);
#endif
    __syncthreads();
  };

  stage(0, 0);
  pipe_sync();
  int buf = 0;
  for (int k0 = 0; k0 < HDIM; k0 += 32, buf ^= 1) {
    if (k0 + 32 < HDIM) stage(k0 + 32, buf ^ 1);
    // hoist all fragments: one load clause, then 8 back-to-back WMMAs
    v16bf bg = lds_frag_b(Bs[buf],      wn * 16 + (lane & 15), lane);
    v16bf bu = lds_frag_b(Bs[buf], 64 + wn * 16 + (lane & 15), lane);
    v16bf a[4];
    #pragma unroll
    for (int mt = 0; mt < 4; ++mt)
      a[mt] = lds_frag_a(As[buf], wm * 64 + mt * 16 + (lane & 15), lane);
    #pragma unroll
    for (int mt = 0; mt < 4; ++mt) {
      accg[mt] = __builtin_amdgcn_wmma_f32_16x16x32_bf16(
          false, a[mt], false, bg, (short)0, accg[mt], false, false);
      accu[mt] = __builtin_amdgcn_wmma_f32_16x16x32_bf16(
          false, a[mt], false, bu, (short)0, accu[mt], false, false);
    }
    pipe_sync();
  }

  const int lhalf = lane >> 4, lcol = lane & 15;
  const int col = fbase + wn * 16 + lcol;
  #pragma unroll
  for (int mt = 0; mt < 4; ++mt) {
    #pragma unroll
    for (int v = 0; v < 8; ++v) {
      int row = tbase + wm * 64 + mt * 16 + lhalf * 8 + v;
      float g = accg[mt][v], u = accu[mt][v];
      float hv = 0.5f * g * (1.0f + erff(g * 0.70710678118654752f)) * u;
      hws[(size_t)row * FDIM + col] = (__bf16)hv;
    }
  }
}

// -------------- GEMM B: out += w_full[:,e] * (h @ Wd_e^T) --------------------
// 128 tokens x 128 h-cols; 8 waves 2(M)x4(N); wave = 4Mx2N 16x16 tiles.
// A-tile (bf16 copy) staged by the Tensor Data Mover when available.
__global__ __launch_bounds__(256)
void moe_down_scale_kernel(const __bf16* __restrict__ hws,   // [T, F] bf16
                           const float*  __restrict__ wd,    // [H, F] f32 (expert e)
                           const float*  __restrict__ wfull, // [T, E] f32
                           float* __restrict__ out,          // [T, H] f32
                           int e)
{
  __shared__ __bf16 As[2][128 * LDS_PITCH];
  __shared__ __bf16 Bs[2][128 * LDS_PITCH];
  const int hbase = blockIdx.x * 128;
  const int tbase = blockIdx.y * 128;
  const int tid  = threadIdx.x;
  const int lane = tid & 31;
  const int w    = tid >> 5;
  const int wm   = w >> 2;
  const int wn   = w & 3;
  const int r    = tid >> 1;
  const int half = tid & 1;

  v8f acc[4][2];
  #pragma unroll
  for (int i = 0; i < 4; ++i)
    #pragma unroll
    for (int n = 0; n < 2; ++n)
      #pragma unroll
      for (int j = 0; j < 8; ++j) acc[i][n][j] = 0.0f;

  const __bf16* atile = hws + (size_t)tbase * FDIM;            // tile origin
  const __bf16* agsrc = hws + (size_t)(tbase + r) * FDIM;      // per-lane row
  const float*  bgsrc = wd + (size_t)(hbase + r) * FDIM;

  auto stage = [&](int k0, int b) {
#if USE_TDM
    if (w == 0) tdm_load_tile_bf16(atile + k0, lds_addr_u32(&As[b][0]));
#else
    copy32B_to_lds(agsrc + k0 + half * 16, &As[b][r * LDS_PITCH + half * 16]);
#endif
    const f32x4* g = (const f32x4*)(bgsrc + k0 + half * 16);
    cvt16_store(g[0], g[1], g[2], g[3], &Bs[b][r * LDS_PITCH + half * 16]);
    if (k0 + 32 < FDIM) __builtin_prefetch(bgsrc + k0 + 32, 0, 1);
  };
  auto pipe_sync = [&]() {
#if USE_TDM
    if (w == 0) __builtin_amdgcn_s_wait_tensorcnt(0);
#elif USE_ASYNC
    __builtin_amdgcn_s_wait_asynccnt(0);
#endif
    __syncthreads();
  };

  stage(0, 0);
  pipe_sync();
  int buf = 0;
  for (int k0 = 0; k0 < FDIM; k0 += 32, buf ^= 1) {
    if (k0 + 32 < FDIM) stage(k0 + 32, buf ^ 1);
    // hoist all fragments: one load clause, then 8 back-to-back WMMAs
    v16bf b0 = lds_frag_b(Bs[buf], wn * 32 +      (lane & 15), lane);
    v16bf b1 = lds_frag_b(Bs[buf], wn * 32 + 16 + (lane & 15), lane);
    v16bf a[4];
    #pragma unroll
    for (int mt = 0; mt < 4; ++mt)
      a[mt] = lds_frag_a(As[buf], wm * 64 + mt * 16 + (lane & 15), lane);
    #pragma unroll
    for (int mt = 0; mt < 4; ++mt) {
      acc[mt][0] = __builtin_amdgcn_wmma_f32_16x16x32_bf16(
          false, a[mt], false, b0, (short)0, acc[mt][0], false, false);
      acc[mt][1] = __builtin_amdgcn_wmma_f32_16x16x32_bf16(
          false, a[mt], false, b1, (short)0, acc[mt][1], false, false);
    }
    pipe_sync();
  }

  const int lhalf = lane >> 4, lcol = lane & 15;
  #pragma unroll
  for (int mt = 0; mt < 4; ++mt) {
    #pragma unroll
    for (int v = 0; v < 8; ++v) {
      int row = tbase + wm * 64 + mt * 16 + lhalf * 8 + v;
      float wgt = wfull[(size_t)row * NEXP + e];
      #pragma unroll
      for (int nt = 0; nt < 2; ++nt) {
        int col = hbase + wn * 32 + nt * 16 + lcol;
        size_t o = (size_t)row * HDIM + col;
        out[o] += wgt * acc[mt][nt][v];
      }
    }
  }
}

// -------------------------------- launcher -----------------------------------
extern "C" void kernel_launch(void* const* d_in, const int* in_sizes, int n_in,
                              void* d_out, int out_size, void* d_ws, size_t ws_size,
                              hipStream_t stream) {
  const float* x   = (const float*)d_in[0];   // [T, H]
  const float* wg  = (const float*)d_in[1];   // [E, H]
  const float* wgu = (const float*)d_in[2];   // [E, 2F, H]
  const float* wd  = (const float*)d_in[3];   // [E, H, F]
  float* out = (float*)d_out;                 // [T, H]

  char* ws = (char*)d_ws;
  __bf16* xb   = (__bf16*)ws;                                             // T*H*2
  __bf16* hws  = (__bf16*)(ws + (size_t)T_TOK * HDIM * 2);                // T*F*2
  float*  wful = (float*)(ws + (size_t)T_TOK * HDIM * 2
                             + (size_t)T_TOK * FDIM * 2);                 // T*E*4

  const int nOut = T_TOK * HDIM;
  zero_f32_kernel<<<(nOut + 255) / 256, 256, 0, stream>>>(out, nOut);

  const int n4 = (T_TOK * HDIM) / 4;
  cvt_f32_bf16_kernel<<<(n4 + 255) / 256, 256, 0, stream>>>(x, xb, n4);

  router_topk_kernel<<<T_TOK / 256, 256, 0, stream>>>(x, wg, wful);

  for (int e = 0; e < NEXP; ++e) {
    moe_gateup_gelu_kernel<<<dim3(FDIM / 64, T_TOK / 128), 256, 0, stream>>>(
        xb, wgu + (size_t)e * 2 * FDIM * HDIM, hws);
    moe_down_scale_kernel<<<dim3(HDIM / 128, T_TOK / 128), 256, 0, stream>>>(
        hws, wd + (size_t)e * HDIM * FDIM, wful, out, e);
  }
}
